// TinyPointNet_11991548690576
// MI455X (gfx1250) — matrix operand order repaired and checked by reading, not compiled, not run on hardware
//
#include <hip/hip_runtime.h>
#include <hip/hip_bf16.h>

typedef _Float16 half_t;
typedef __attribute__((ext_vector_type(16))) _Float16 v16h;
typedef __attribute__((ext_vector_type(8)))  float    v8f;

union HF16 { v16h h; int4 i2[2]; };
union HF8  { _Float16 h[8]; int4 i; };
union HPK  { _Float16 hh[2]; unsigned uu; };

// monotone float<->uint encoding so atomicMax/Min on unsigned == float max/min
__device__ __forceinline__ unsigned fenc(float f){
  unsigned u = __float_as_uint(f);
  return (u & 0x80000000u) ? ~u : (u | 0x80000000u);
}
__device__ __forceinline__ float fdec(unsigned u){
  unsigned v = (u & 0x80000000u) ? (u & 0x7FFFFFFFu) : ~u;
  return __uint_as_float(v);
}

// full-wave sum via ds_swizzle xor patterns (SWAPX1/2/4/8/16) -- no guard code
__device__ __forceinline__ float xsum32(float v){
  v += __int_as_float(__builtin_amdgcn_ds_swizzle(__float_as_int(v), 0x041F));
  v += __int_as_float(__builtin_amdgcn_ds_swizzle(__float_as_int(v), 0x081F));
  v += __int_as_float(__builtin_amdgcn_ds_swizzle(__float_as_int(v), 0x101F));
  v += __int_as_float(__builtin_amdgcn_ds_swizzle(__float_as_int(v), 0x201F));
  v += __int_as_float(__builtin_amdgcn_ds_swizzle(__float_as_int(v), 0x401F));
  return v;
}

// CDNA5 async global->LDS copy (ASYNCcnt-tracked); lds_off = low 32 bits of generic shared ptr
__device__ __forceinline__ void async_ld16(unsigned lds_off, const half_t* g){
  asm volatile("global_load_async_to_lds_b128 %0, %1, off"
               :: "v"(lds_off), "v"((unsigned long long)(size_t)g) : "memory");
}
__device__ __forceinline__ void wait_async0(){
  asm volatile("s_wait_asynccnt 0" ::: "memory");
}

// ---------------------------------------------------------------- utilities
__global__ void kfillf(float* p, int n, float v){
  int i = blockIdx.x*256 + threadIdx.x; if (i < n) p[i] = v;
}
__global__ void kfillu(unsigned* p, int n, unsigned v){
  int i = blockIdx.x*256 + threadIdx.x; if (i < n) p[i] = v;
}
__global__ void kw2h(const float* __restrict__ w, half_t* __restrict__ o, int n){
  int i = blockIdx.x*256 + threadIdx.x; if (i < n) o[i] = (half_t)w[i];
}
// BN affine from accumulated stats: next-layer input = relu(s*y + t)
__global__ void kstats(const float* __restrict__ ssum, const float* __restrict__ ssq,
                       const float* __restrict__ g, const float* __restrict__ be,
                       float* __restrict__ sc, float* __restrict__ sh, int C, float cnt){
  int c = blockIdx.x*64 + threadIdx.x;
  if (c < C){
    float m = ssum[c] / cnt;
    float v = ssq[c] / cnt - m*m;
    float s = g[c] * rsqrtf(v + 1e-5f);
    sc[c] = s; sh[c] = be[c] - m*s;
  }
}
// max_n relu(s*y+t) == relu(s*max(y)+t) for s>=0 else relu(s*min(y)+t)
__global__ void kflat(const unsigned* __restrict__ emax, const unsigned* __restrict__ emin,
                      const float* __restrict__ sc, const float* __restrict__ sh,
                      float* __restrict__ out, int C){
  int i = blockIdx.x*256 + threadIdx.x;
  if (i < 32*C){
    int c = i % C;
    float s = sc[c], t = sh[c];
    float v = (s >= 0.f) ? fdec(emax[i]) : fdec(emin[i]);
    out[i] = fmaxf(s*v + t, 0.f);
  }
}

// ---------------------------------------------------------------- SHOT rotation
__global__ __launch_bounds__(256) void krot(const float* __restrict__ x, float* __restrict__ R){
  int b = blockIdx.x, tid = threadIdx.x;
  const float* xb = x + (size_t)b*3*8192;
  __shared__ float red[256];
  __shared__ int   redi[256];
  __shared__ float mean[3];
  __shared__ float covs[7];
  __shared__ float V[9];
  __shared__ int   posTot[3];

  for (int c = 0; c < 3; ++c){
    float s = 0.f;
    for (int n = tid; n < 8192; n += 256) s += xb[c*8192 + n];
    red[tid] = s; __syncthreads();
    for (int o = 128; o > 0; o >>= 1){ if (tid < o) red[tid] += red[tid+o]; __syncthreads(); }
    if (tid == 0) mean[c] = red[0] * (1.f/8192.f);
    __syncthreads();
  }
  float acc[7] = {0,0,0,0,0,0,0};
  for (int n = tid; n < 8192; n += 256){
    float c0 = xb[n] - mean[0], c1 = xb[8192+n] - mean[1], c2 = xb[16384+n] - mean[2];
    float d = sqrtf(c0*c0 + c1*c1 + c2*c2);
    float w = fmaxf(1.f - d, 0.f);
    acc[0] += w*c0*c0; acc[1] += w*c0*c1; acc[2] += w*c0*c2;
    acc[3] += w*c1*c1; acc[4] += w*c1*c2; acc[5] += w*c2*c2; acc[6] += w;
  }
  for (int i = 0; i < 7; ++i){
    red[tid] = acc[i]; __syncthreads();
    for (int o = 128; o > 0; o >>= 1){ if (tid < o) red[tid] += red[tid+o]; __syncthreads(); }
    if (tid == 0) covs[i] = red[0];
    __syncthreads();
  }
  if (tid == 0){
    float inv = covs[6] > 0.f ? 1.f/covs[6] : 1.f;
    float a00 = covs[0]*inv + 1e-6f, a01 = covs[1]*inv, a02 = covs[2]*inv;
    float a11 = covs[3]*inv + 1e-6f, a12 = covs[4]*inv, a22 = covs[5]*inv + 1e-6f;
    float q  = (a00 + a11 + a22) * (1.f/3.f);
    float p1 = a01*a01 + a02*a02 + a12*a12;
    float b00 = a00-q, b11 = a11-q, b22 = a22-q;
    float p2 = b00*b00 + b11*b11 + b22*b22 + 2.f*p1;
    float ev[3];
    if (p2 < 1e-22f){ ev[0] = ev[1] = ev[2] = q; }
    else {
      float p  = sqrtf(p2 * (1.f/6.f));
      float ip = 1.f/p;
      float c00=b00*ip, c01=a01*ip, c02=a02*ip, c11=b11*ip, c12=a12*ip, c22=b22*ip;
      float detB = c00*(c11*c22 - c12*c12) - c01*(c01*c22 - c12*c02) + c02*(c01*c12 - c11*c02);
      float r   = fminf(fmaxf(detB*0.5f, -1.f), 1.f);
      float phi = acosf(r) * (1.f/3.f);
      ev[0] = q + 2.f*p*cosf(phi);
      ev[2] = q + 2.f*p*cosf(phi + 2.0943951023931953f);
      ev[1] = 3.f*q - ev[0] - ev[2];
    }
    for (int k = 0; k < 3; ++k){       // columns descending (vecs[:,::-1])
      float m00 = a00-ev[k], m11 = a11-ev[k], m22 = a22-ev[k];
      float v0x = a01*a12 - a02*m11, v0y = a02*a01 - m00*a12, v0z = m00*m11 - a01*a01;
      float v1x = a01*m22 - a02*a12, v1y = a02*a02 - m00*m22, v1z = m00*a12 - a01*a02;
      float v2x = m11*m22 - a12*a12, v2y = a12*a02 - a01*m22, v2z = a01*a12 - m11*a02;
      float n0 = v0x*v0x+v0y*v0y+v0z*v0z;
      float n1 = v1x*v1x+v1y*v1y+v1z*v1z;
      float n2 = v2x*v2x+v2y*v2y+v2z*v2z;
      float vx=v0x, vy=v0y, vz=v0z, nn=n0;
      if (n1 > nn){ vx=v1x; vy=v1y; vz=v1z; nn=n1; }
      if (n2 > nn){ vx=v2x; vy=v2y; vz=v2z; nn=n2; }
      if (nn < 1e-24f){ vx = (k==0); vy = (k==1); vz = (k==2); nn = 1.f; }
      float rn = rsqrtf(nn);
      V[0*3+k] = vx*rn; V[1*3+k] = vy*rn; V[2*3+k] = vz*rn;
    }
  }
  __syncthreads();
  int pos[3] = {0,0,0};
  for (int n = tid; n < 8192; n += 256){
    float c0 = xb[n] - mean[0], c1 = xb[8192+n] - mean[1], c2 = xb[16384+n] - mean[2];
    for (int k = 0; k < 3; ++k){
      float pr = c0*V[k] + c1*V[3+k] + c2*V[6+k];
      pos[k] += (pr >= 0.f) ? 1 : 0;
    }
  }
  for (int k = 0; k < 3; ++k){
    redi[tid] = pos[k]; __syncthreads();
    for (int o = 128; o > 0; o >>= 1){ if (tid < o) redi[tid] += redi[tid+o]; __syncthreads(); }
    if (tid == 0) posTot[k] = redi[0];
    __syncthreads();
  }
  if (tid == 0){
    for (int k = 0; k < 3; ++k){
      float fl = (2*posTot[k] < 8192) ? -1.f : 1.f;
      for (int c = 0; c < 3; ++c) R[b*9 + c*3 + k] = V[c*3+k] * fl;
    }
  }
}

// ------------------------------------------- m1: rotation folded into 64x3 weights (K=3 -> scalar)
__global__ __launch_bounds__(256) void km1(const float* __restrict__ x, const float* __restrict__ R,
                                           const float* __restrict__ w1, const float* __restrict__ b1,
                                           half_t* __restrict__ y1,
                                           float* __restrict__ ssum, float* __restrict__ ssq){
  int b = blockIdx.x >> 5;                       // 32 blocks per batch
  int n = ((blockIdx.x & 31) << 8) + threadIdx.x;
  int tid = threadIdx.x;
  int lane = tid & 31;
  __shared__ float rs[9];
  __shared__ float We[64][3];
  __shared__ float bsum[64], bsq[64];
  if (tid < 9)  rs[tid] = R[b*9 + tid];
  if (tid < 64){ bsum[tid] = 0.f; bsq[tid] = 0.f; }
  __syncthreads();
  if (tid < 64){
    int o = tid;
    for (int c = 0; c < 3; ++c)
      We[o][c] = w1[o*3+0]*rs[c*3+0] + w1[o*3+1]*rs[c*3+1] + w1[o*3+2]*rs[c*3+2];
  }
  __syncthreads();
  const float* xb = x + (size_t)b*3*8192;
  float x0 = xb[n], x1 = xb[8192+n], x2 = xb[16384+n];
  half_t* yb = y1 + (size_t)b*64*8192 + n;
  for (int o = 0; o < 64; ++o){
    float y = We[o][0]*x0 + We[o][1]*x1 + We[o][2]*x2 + b1[o];
    yb[(size_t)o*8192] = (half_t)y;
    float s1 = xsum32(y), s2 = xsum32(y*y);   // ds_swizzle wave-reduce, 1 ds_add per wave
    if (lane == 0){ atomicAdd(&bsum[o], s1); atomicAdd(&bsq[o], s2); }
  }
  __syncthreads();
  if (tid < 64){ atomicAdd(&ssum[tid], bsum[tid]); atomicAdd(&ssq[tid], bsq[tid]); }
}

// ------------------------------------------- WMMA GEMM, all behavior flags compile-time
// D[b,co,n] = sum_k W[co,k] * relu(in_s[k]*X[b,k,n]+in_t[k]) + bias[co]
// Block tile: 64 Cout x 64 points; 8 waves = 2 Cout tiles x 4 point tiles, 2 acc/wave.
// Weights: async global->LDS DMA (overlaps X staging), A-frags via ds_load_b128.
// Stats: slotted LDS accumulators [64 rows][16 cols] -- each lane atomically adds to
// its own slot (consecutive banks, no intra-instruction serialization; 4-way cross-wave).
template<int CIN, bool AFF, bool HASY, bool STATS, bool MM, bool BW>
__global__ __launch_bounds__(256) void gemm_bn(
    const half_t* __restrict__ X, const float* __restrict__ in_s, const float* __restrict__ in_t,
    const half_t* __restrict__ W, const float* __restrict__ bias,
    half_t* __restrict__ Y, float* __restrict__ ssum, float* __restrict__ ssq,
    unsigned* __restrict__ emax, unsigned* __restrict__ emin, int Cout)
{
  const int N = 8192;
  constexpr int NCH = CIN / 32;
  int b  = blockIdx.x >> 7;                 // 128 blocks of 64 points per batch
  int n0 = (blockIdx.x & 127) * 64;
  int coBase = blockIdx.y * 64;
  int tid  = threadIdx.x;
  int lane = tid & 31, wave = tid >> 5;     // wave32
  int cot = wave & 1, ptt = wave >> 1;
  int hi = lane >> 4, lo = lane & 15;

  __shared__ half_t Xs[64][CIN + 8];        // [point][K] transposed input tile
  __shared__ half_t Ws[64][CIN + 8];        // [co][K] weight tile (async-DMA filled)
  __shared__ float    rsum[STATS ? 1024 : 1], rsq[STATS ? 1024 : 1];
  __shared__ unsigned rmax[MM ? 1024 : 1],   rmin[MM ? 1024 : 1];
  if (STATS) for (int i = tid; i < 1024; i += 256){ rsum[i] = 0.f; rsq[i] = 0.f; }
  if (MM)    for (int i = tid; i < 1024; i += 256){ rmax[i] = 0x007FFFFFu; rmin[i] = 0xFF800000u; }

  const half_t* Wb = W + (BW ? (size_t)b*(size_t)Cout*CIN : 0);
  const half_t* Xb = X + (size_t)b*CIN*N;

  // ---- kick off async weight DMA first so it overlaps the X transform below
  {
    constexpr int CHUNKS = 64*CIN/8;        // 16B chunks
    #pragma unroll
    for (int ch = 0; ch < CHUNKS/256; ++ch){
      int idx = ch*256 + tid;
      int row = idx / (CIN/8);
      int col = (idx % (CIN/8)) * 8;        // halves
      async_ld16((unsigned)(size_t)&Ws[row][col],
                 Wb + (size_t)(coBase + row)*CIN + col);
    }
  }

  // ---- stage input tile: row pair per thread -> packed b32 LDS stores
  {
    int c2 = tid >> 3;                      // 0..31 -> rows (2*c2, 2*c2+1)
    int p0 = (tid & 7) * 8;                 // 8 consecutive points
    #pragma unroll
    for (int h = 0; h < CIN/64; ++h){
      int c = h*64 + 2*c2;
      const half_t* s0 = Xb + (size_t)c*N + n0 + p0;
      const half_t* s1 = s0 + N;
      __builtin_prefetch(s0 + 64, 0, 1);    // next point tile (gfx1250 global_prefetch)
      HF8 u0, u1; u0.i = *(const int4*)s0; u1.i = *(const int4*)s1;
      float sa = 1.f, ta = 0.f, sb = 1.f, tb = 0.f;
      if (AFF){ sa = in_s[c]; ta = in_t[c]; sb = in_s[c+1]; tb = in_t[c+1]; }
      #pragma unroll
      for (int j = 0; j < 8; ++j){
        float va = (float)u0.h[j], vb = (float)u1.h[j];
        if (AFF){ va = fmaxf(va*sa + ta, 0.f); vb = fmaxf(vb*sb + tb, 0.f); }
        HPK pk; pk.hh[0] = (half_t)va; pk.hh[1] = (half_t)vb;
        *(unsigned*)&Xs[p0 + j][c] = pk.uu;
      }
    }
  }
  wait_async0();                            // s_wait_asynccnt 0 (our wave's DMA landed)
  __syncthreads();                          // all waves' stores + DMA visible

  // ---- A fragments from LDS (ISA 16-bit A 16x32 layout)
  HF16 afr[2][NCH];
  {
    int r0 = cot*32 + lo;
    #pragma unroll
    for (int q = 0; q < NCH; ++q){
      const half_t* wp0 = &Ws[r0][q*32 + hi*8];
      const half_t* wp1 = &Ws[r0 + 16][q*32 + hi*8];
      afr[0][q].i2[0] = *(const int4*)wp0;
      afr[0][q].i2[1] = *(const int4*)(wp0 + 16);
      afr[1][q].i2[0] = *(const int4*)wp1;
      afr[1][q].i2[1] = *(const int4*)(wp1 + 16);
    }
  }

  // ---- MAC loop: fully unrolled, 2 WMMAs per B fragment
  v8f acc0 = {}, acc1 = {};
  #pragma unroll
  for (int q = 0; q < NCH; ++q){
    HF16 bfr;
    const half_t* xp = &Xs[ptt*16 + lo][q*32 + hi*8];
    bfr.i2[0] = *(const int4*)xp;
    bfr.i2[1] = *(const int4*)(xp + 16);
    acc0 = __builtin_amdgcn_wmma_f32_16x16x32_f16(false, afr[0][q].h, false, bfr.h,
                                                  (short)0, acc0, false, false);
    acc1 = __builtin_amdgcn_wmma_f32_16x16x32_f16(false, afr[1][q].h, false, bfr.h,
                                                  (short)0, acc1, false, false);
  }

  // ---- epilogue: bias, optional store, slotted stats / max-min
  #pragma unroll
  for (int hf = 0; hf < 2; ++hf){
    v8f a = hf ? acc1 : acc0;
    #pragma unroll
    for (int v = 0; v < 8; ++v){            // C/D layout: m = v + 8*hi, col = lo
      int m   = v + hi*8;
      int lco = cot*32 + hf*16 + m;
      int co  = coBase + lco;
      int pt  = n0 + ptt*16 + lo;
      float val = a[v];
      if (!BW) val += bias[co];
      if (HASY) Y[((size_t)b*Cout + co)*N + pt] = (half_t)val;
      if (STATS){ atomicAdd(&rsum[lco*16 + lo], val); atomicAdd(&rsq[lco*16 + lo], val*val); }
      if (MM){ unsigned e = fenc(val);
               atomicMax(&rmax[lco*16 + lo], e); atomicMin(&rmin[lco*16 + lo], e); }
    }
  }
  if (STATS || MM){
    __syncthreads();
    if (tid < 64){
      int co = coBase + tid;
      if (STATS){
        float s = 0.f, q = 0.f;
        #pragma unroll
        for (int j = 0; j < 16; ++j){ s += rsum[tid*16 + j]; q += rsq[tid*16 + j]; }
        atomicAdd(&ssum[co], s); atomicAdd(&ssq[co], q);
      }
      if (MM){
        unsigned mx = 0x007FFFFFu, mn = 0xFF800000u;
        #pragma unroll
        for (int j = 0; j < 16; ++j){
          unsigned a1 = rmax[tid*16 + j]; mx = (a1 > mx) ? a1 : mx;
          unsigned a2 = rmin[tid*16 + j]; mn = (a2 < mn) ? a2 : mn;
        }
        atomicMax(&emax[(size_t)b*Cout + co], mx);
        atomicMin(&emin[(size_t)b*Cout + co], mn);
      }
    }
  }
}

// ------------------------------------------- small FC + BN over batch (32 rows)
__global__ __launch_bounds__(256) void kfc(const float* __restrict__ Xin, const float* __restrict__ w,
                                           const float* __restrict__ bias, const float* __restrict__ g,
                                           const float* __restrict__ be, float* __restrict__ out, int K){
  int co = blockIdx.x;
  int tid = threadIdx.x;
  int grp = tid >> 3, lt = tid & 7;         // 32 batches x 8-thread dot groups
  __shared__ float partial[256];
  __shared__ float ycol[32];
  __shared__ float mv[2];
  const float* xr = Xin + (size_t)grp*K;
  const float* wr = w   + (size_t)co*K;
  float s = 0.f;
  for (int k = lt; k < K; k += 8) s += xr[k]*wr[k];
  partial[tid] = s; __syncthreads();
  if (lt == 0){
    float t = 0.f;
    for (int j = 0; j < 8; ++j) t += partial[grp*8 + j];
    ycol[grp] = t + bias[co];
  }
  __syncthreads();
  if (tid == 0){
    float m = 0.f;
    for (int j = 0; j < 32; ++j) m += ycol[j];
    m *= (1.f/32.f);
    float v = 0.f;
    for (int j = 0; j < 32; ++j){ float d = ycol[j]-m; v += d*d; }
    v *= (1.f/32.f);
    mv[0] = m; mv[1] = rsqrtf(v + 1e-5f);
  }
  __syncthreads();
  if (tid < 32)
    out[(size_t)tid*gridDim.x + co] = fmaxf(g[co]*((ycol[tid]-mv[0])*mv[1]) + be[co], 0.f);
}

// ------------------------------------------- fc3 -> per-batch feature transform ft^T + I, f16
__global__ __launch_bounds__(256) void kfc3(const float* __restrict__ h2, const float* __restrict__ w,
                                            const float* __restrict__ bias, half_t* __restrict__ wft){
  int i = blockIdx.x*256 + threadIdx.x;     // [b][k][c]
  int b = i >> 12;
  int j = i & 4095;
  int k = j >> 6, c = j & 63;
  int o = c*64 + k;                         // fc3 output index (ft[b][c][k])
  const float* hr = h2 + (size_t)b*256;
  const float* wr = w  + (size_t)o*256;
  float s = 0.f;
  for (int q = 0; q < 256; ++q) s += hr[q]*wr[q];
  s += bias[o] + ((c == k) ? 1.f : 0.f);
  wft[i] = (half_t)s;
}

extern "C" void kernel_launch(void* const* d_in, const int* in_sizes, int n_in,
                              void* d_out, int out_size, void* d_ws, size_t ws_size,
                              hipStream_t stream){
  (void)in_sizes; (void)n_in; (void)out_size; (void)ws_size;
  const float* x    = (const float*)d_in[0];
  const float* m1w  = (const float*)d_in[1],  *m1b  = (const float*)d_in[2],  *m1g  = (const float*)d_in[3],  *m1be = (const float*)d_in[4];
  const float* m2w  = (const float*)d_in[5],  *m2b  = (const float*)d_in[6],  *m2g  = (const float*)d_in[7],  *m2be = (const float*)d_in[8];
  const float* m3w  = (const float*)d_in[9],  *m3b  = (const float*)d_in[10], *m3g  = (const float*)d_in[11], *m3be = (const float*)d_in[12];
  const float* m4w  = (const float*)d_in[13], *m4b  = (const float*)d_in[14], *m4g  = (const float*)d_in[15], *m4be = (const float*)d_in[16];
  const float* m5w  = (const float*)d_in[17], *m5b  = (const float*)d_in[18], *m5g  = (const float*)d_in[19], *m5be = (const float*)d_in[20];
  const float* t1w  = (const float*)d_in[21], *t1b  = (const float*)d_in[22], *t1g  = (const float*)d_in[23], *t1be = (const float*)d_in[24];
  const float* t2w  = (const float*)d_in[25], *t2b  = (const float*)d_in[26], *t2g  = (const float*)d_in[27], *t2be = (const float*)d_in[28];
  const float* t3w  = (const float*)d_in[29], *t3b  = (const float*)d_in[30], *t3g  = (const float*)d_in[31], *t3be = (const float*)d_in[32];
  const float* tf1w = (const float*)d_in[33], *tf1b = (const float*)d_in[34], *tf1g = (const float*)d_in[35], *tf1be= (const float*)d_in[36];
  const float* tf2w = (const float*)d_in[37], *tf2b = (const float*)d_in[38], *tf2g = (const float*)d_in[39], *tf2be= (const float*)d_in[40];
  const float* fc3w = (const float*)d_in[41], *fc3b = (const float*)d_in[42];

  char* ws = (char*)d_ws;
  size_t off = 0;
  auto A = [&](size_t bytes)->char*{ char* p = ws + off; off += (bytes + 255) & ~(size_t)255; return p; };

  float*    R      = (float*)A(32*9*4);
  float*    stats  = (float*)A(3584*4);         // sums & sumsqs
  float*    sst    = (float*)A(3584*4);         // scales & shifts
  unsigned* emax3  = (unsigned*)A(32*1024*4);
  unsigned* emin3  = (unsigned*)A(32*1024*4);
  unsigned* emax5  = (unsigned*)A(32*256*4);
  unsigned* emin5  = (unsigned*)A(32*256*4);
  float*    flat   = (float*)A(32*1024*4);
  float*    h1     = (float*)A(32*512*4);
  float*    h2     = (float*)A(32*256*4);
  half_t*   wft    = (half_t*)A((size_t)32*64*64*2);
  half_t*   hw_m2  = (half_t*)A(64*64*2);
  half_t*   hw_t1  = (half_t*)A(64*64*2);
  half_t*   hw_t2  = (half_t*)A(128*64*2);
  half_t*   hw_t3  = (half_t*)A((size_t)1024*128*2);
  half_t*   hw_m3  = (half_t*)A(64*64*2);
  half_t*   hw_m4  = (half_t*)A(128*64*2);
  half_t*   hw_m5  = (half_t*)A(256*128*2);
  half_t*   bufA   = (half_t*)A((size_t)32*64*8192*2);    // y1 -> yt1 -> y3
  half_t*   bufB   = (half_t*)A((size_t)32*64*8192*2);    // y2 (a2 raw, used twice)
  half_t*   bufC   = (half_t*)A((size_t)32*128*8192*2);   // yt2 -> a2t -> y4

  float* sum_m1 = stats;          float* sq_m1 = sum_m1 + 64;
  float* sum_m2 = sq_m1  + 64;    float* sq_m2 = sum_m2 + 64;
  float* sum_t1 = sq_m2  + 64;    float* sq_t1 = sum_t1 + 64;
  float* sum_t2 = sq_t1  + 64;    float* sq_t2 = sum_t2 + 128;
  float* sum_t3 = sq_t2  + 128;   float* sq_t3 = sum_t3 + 1024;
  float* sum_m3 = sq_t3  + 1024;  float* sq_m3 = sum_m3 + 64;
  float* sum_m4 = sq_m3  + 64;    float* sq_m4 = sum_m4 + 128;
  float* sum_m5 = sq_m4  + 128;   float* sq_m5 = sum_m5 + 256;

  float* sc_m1 = sst;             float* sh_m1 = sc_m1 + 64;
  float* sc_m2 = sh_m1 + 64;      float* sh_m2 = sc_m2 + 64;
  float* sc_t1 = sh_m2 + 64;      float* sh_t1 = sc_t1 + 64;
  float* sc_t2 = sh_t1 + 64;      float* sh_t2 = sc_t2 + 128;
  float* sc_t3 = sh_t2 + 128;     float* sh_t3 = sc_t3 + 1024;
  float* sc_m3 = sh_t3 + 1024;    float* sh_m3 = sc_m3 + 64;
  float* sc_m4 = sh_m3 + 64;      float* sh_m4 = sc_m4 + 128;
  float* sc_m5 = sh_m4 + 128;     float* sh_m5 = sc_m5 + 256;

  const float CNT = 262144.f;     // B*N

  // init accumulators (every call; capture-safe, no memset/malloc)
  kfillf<<<14, 256, 0, stream>>>(stats, 3584, 0.f);
  kfillu<<<128, 256, 0, stream>>>(emax3, 32768, 0x007FFFFFu);  // enc(-inf)
  kfillu<<<128, 256, 0, stream>>>(emin3, 32768, 0xFF800000u);  // enc(+inf)
  kfillu<<<32, 256, 0, stream>>>(emax5, 8192, 0x007FFFFFu);
  kfillu<<<32, 256, 0, stream>>>(emin5, 8192, 0xFF800000u);

  // weights -> f16
  kw2h<<<16, 256, 0, stream>>>(m2w, hw_m2, 4096);
  kw2h<<<16, 256, 0, stream>>>(t1w, hw_t1, 4096);
  kw2h<<<32, 256, 0, stream>>>(t2w, hw_t2, 8192);
  kw2h<<<512, 256, 0, stream>>>(t3w, hw_t3, 131072);
  kw2h<<<16, 256, 0, stream>>>(m3w, hw_m3, 4096);
  kw2h<<<32, 256, 0, stream>>>(m4w, hw_m4, 8192);
  kw2h<<<128, 256, 0, stream>>>(m5w, hw_m5, 32768);

  // SHOT rotation + m1 (rotation folded into weights)
  krot<<<32, 256, 0, stream>>>(x, R);
  km1<<<1024, 256, 0, stream>>>(x, R, m1w, m1b, bufA, sum_m1, sq_m1);
  kstats<<<1, 64, 0, stream>>>(sum_m1, sq_m1, m1g, m1be, sc_m1, sh_m1, 64, CNT);

  // m2
  gemm_bn<64,true,true,true,false,false><<<dim3(4096,1), 256, 0, stream>>>(
      bufA, sc_m1, sh_m1, hw_m2, m2b, bufB, sum_m2, sq_m2, nullptr, nullptr, 64);
  kstats<<<1, 64, 0, stream>>>(sum_m2, sq_m2, m2g, m2be, sc_m2, sh_m2, 64, CNT);

  // tnet: t1, t2, t3 (t3 never materialized; stats + per-(b,ch) max/min only)
  gemm_bn<64,true,true,true,false,false><<<dim3(4096,1), 256, 0, stream>>>(
      bufB, sc_m2, sh_m2, hw_t1, t1b, bufA, sum_t1, sq_t1, nullptr, nullptr, 64);
  kstats<<<1, 64, 0, stream>>>(sum_t1, sq_t1, t1g, t1be, sc_t1, sh_t1, 64, CNT);
  gemm_bn<64,true,true,true,false,false><<<dim3(4096,2), 256, 0, stream>>>(
      bufA, sc_t1, sh_t1, hw_t2, t2b, bufC, sum_t2, sq_t2, nullptr, nullptr, 128);
  kstats<<<2, 64, 0, stream>>>(sum_t2, sq_t2, t2g, t2be, sc_t2, sh_t2, 128, CNT);
  gemm_bn<128,true,false,true,true,false><<<dim3(4096,16), 256, 0, stream>>>(
      bufC, sc_t2, sh_t2, hw_t3, t3b, nullptr, sum_t3, sq_t3, emax3, emin3, 1024);
  kstats<<<16, 64, 0, stream>>>(sum_t3, sq_t3, t3g, t3be, sc_t3, sh_t3, 1024, CNT);
  kflat<<<128, 256, 0, stream>>>(emax3, emin3, sc_t3, sh_t3, flat, 1024);

  // tnet FC head
  kfc<<<512, 256, 0, stream>>>(flat, tf1w, tf1b, tf1g, tf1be, h1, 1024);
  kfc<<<256, 256, 0, stream>>>(h1, tf2w, tf2b, tf2g, tf2be, h2, 512);
  kfc3<<<512, 256, 0, stream>>>(h2, fc3w, fc3b, wft);

  // apply per-batch feature transform to a2 (batched weights, raw output, no bias)
  gemm_bn<64,true,true,false,false,true><<<dim3(4096,1), 256, 0, stream>>>(
      bufB, sc_m2, sh_m2, wft, nullptr, bufC, nullptr, nullptr, nullptr, nullptr, 64);

  // m3, m4, m5 (m5 never materialized)
  gemm_bn<64,false,true,true,false,false><<<dim3(4096,1), 256, 0, stream>>>(
      bufC, nullptr, nullptr, hw_m3, m3b, bufA, sum_m3, sq_m3, nullptr, nullptr, 64);
  kstats<<<1, 64, 0, stream>>>(sum_m3, sq_m3, m3g, m3be, sc_m3, sh_m3, 64, CNT);
  gemm_bn<64,true,true,true,false,false><<<dim3(4096,2), 256, 0, stream>>>(
      bufA, sc_m3, sh_m3, hw_m4, m4b, bufC, sum_m4, sq_m4, nullptr, nullptr, 128);
  kstats<<<2, 64, 0, stream>>>(sum_m4, sq_m4, m4g, m4be, sc_m4, sh_m4, 128, CNT);
  gemm_bn<128,true,false,true,true,false><<<dim3(4096,4), 256, 0, stream>>>(
      bufC, sc_m4, sh_m4, hw_m5, m5b, nullptr, sum_m5, sq_m5, emax5, emin5, 256);
  kstats<<<4, 64, 0, stream>>>(sum_m5, sq_m5, m5g, m5be, sc_m5, sh_m5, 256, CNT);

  // final: out[b][c] = relu(affine(max/min over n))
  kflat<<<32, 256, 0, stream>>>(emax5, emin5, sc_m5, sh_m5, (float*)d_out, 256);
}
